// MyNet_88725434401248
// MI455X (gfx1250) — matrix operand order repaired and checked by reading, not compiled, not run on hardware
//
#include <hip/hip_runtime.h>

typedef _Float16 f16;
typedef _Float16 v16h __attribute__((ext_vector_type(16)));
typedef float    v8f  __attribute__((ext_vector_type(8)));

#define B_    2
#define S_    336
#define V_    8
#define D_    256
#define H_    8
#define DK_   32
#define P_    96
#define NTOK  5376            // B*V*S == B*S*V
#define KD    (S_ * D_)       // 86016
#define LN_EPS 1e-5f
#define SCALE  0.1767766952966369f   // 1/sqrt(32)

// ---------------- WMMA fragment helpers (gfx1250 wave32 layouts) ----------------
// A 16x32 f16: lane<16 row=lane holds K {k0..k0+7, k0+16..k0+23};
//              lane>=16 row=lane-16 holds K {k0+8..k0+15, k0+24..k0+31}.
__device__ __forceinline__ v16h load_a_f32(const float* src, int ld, int row0, int k0) {
  int lane = threadIdx.x & 31;
  int m  = row0 + (lane & 15);
  int kb = k0 + ((lane >> 4) << 3);
  const float* p = src + (size_t)m * ld + kb;
  v16h a;
#pragma unroll
  for (int e = 0; e < 8; ++e) a[e] = (f16)p[e];
#pragma unroll
  for (int e = 0; e < 8; ++e) a[8 + e] = (f16)p[16 + e];
  return a;
}

__device__ __forceinline__ v16h load_a_f16(const f16* src, int ld, int row0, int k0) {
  int lane = threadIdx.x & 31;
  int m  = row0 + (lane & 15);
  int kb = k0 + ((lane >> 4) << 3);
  const f16* p = src + (size_t)m * ld + kb;
  v16h a;
#pragma unroll
  for (int e = 0; e < 8; ++e) a[e] = p[e];
#pragma unroll
  for (int e = 0; e < 8; ++e) a[8 + e] = p[16 + e];
  return a;
}

// B 32x16 f16 from transposed storage wt[n][k]: lane<16 col=lane K {k0..k0+7, k0+16..k0+23};
// lane>=16 col=lane-16 K {k0+8..k0+15, k0+24..k0+31}.
__device__ __forceinline__ v16h load_b_f16(const f16* wt, int ldk, int n0, int k0) {
  int lane = threadIdx.x & 31;
  int n  = n0 + (lane & 15);
  int kb = k0 + ((lane >> 4) << 3);
  const f16* p = wt + (size_t)n * ldk + kb;
  v16h b;
#pragma unroll
  for (int e = 0; e < 8; ++e) b[e] = p[e];
#pragma unroll
  for (int e = 0; e < 8; ++e) b[8 + e] = p[16 + e];
  return b;
}

__device__ __forceinline__ v8f wmma16(v16h a, v16h b, v8f c) {
  return __builtin_amdgcn_wmma_f32_16x16x32_f16(false, a, false, b, (short)0, c, false, false);
}

// ---------------- prep kernels ----------------
__global__ void k_embed(const float* __restrict__ x, const float* __restrict__ ew,
                        const float* __restrict__ eb, float* __restrict__ e1,
                        f16* __restrict__ e1h) {
  int t = blockIdx.x, d = threadIdx.x;              // token in [B,V,S] order
  int b = t / (V_ * S_);
  int rem = t - b * V_ * S_;
  int v = rem / S_;
  int s = rem - v * S_;
  float xv = x[((size_t)b * S_ + s) * V_ + v];
  float val = xv * ew[d] + eb[d];
  e1[(size_t)t * D_ + d] = val;
  e1h[(size_t)t * D_ + d] = (f16)val;
}

__global__ void k_wtrans(const float* __restrict__ w, f16* __restrict__ wt) {
  int n = blockIdx.x, k = threadIdx.x;              // wt[n][k] = W[k][n]
  wt[(size_t)n * D_ + k] = (f16)w[(size_t)k * D_ + n];
}

__global__ void k_pwtrans(const float* __restrict__ pw, f16* __restrict__ pwt) {
  int k = blockIdx.x * blockDim.x + threadIdx.x;    // 0..86015
  for (int p = 0; p < P_; ++p)
    pwt[(size_t)p * KD + k] = (f16)pw[(size_t)k * P_ + p];
}

// ---------------- QKV projection (WMMA) ----------------
// 128 threads / 4 waves per 16-token tile; each wave hoists the A fragments once
// and sweeps 12 of the 48 (matrix z, n-tile) combos.
// mode 0 (time axis): V stored transposed per group of 336 -> vt[g][d][l]
// mode 1 (var axis) : V stored row-major vb[tok][d]
__global__ void k_qkv(const f16* __restrict__ eh,
                      const f16* __restrict__ wq, const f16* __restrict__ wk, const f16* __restrict__ wv,
                      const float* __restrict__ bq, const float* __restrict__ bk, const float* __restrict__ bv,
                      f16* __restrict__ qb, f16* __restrict__ kb, f16* __restrict__ vb,
                      f16* __restrict__ vt, int mode) {
  int m0 = blockIdx.x * 16;
  int wave = threadIdx.x >> 5, lane = threadIdx.x & 31;
  v16h a[8];
#pragma unroll
  for (int k = 0; k < 8; ++k) a[k] = load_a_f16(eh, D_, m0, k * 32);
  int rbase = (lane < 16) ? 0 : 8;
  int lc = lane & 15;
  for (int i = 0; i < 12; ++i) {
    int c = wave * 12 + i;                 // 0..47
    int z = c >> 4, n0 = (c & 15) * 16;
    const f16*   wt   = (z == 0) ? wq : (z == 1) ? wk : wv;
    const float* bias = (z == 0) ? bq : (z == 1) ? bk : bv;
    v8f acc = {};
#pragma unroll
    for (int k = 0; k < 8; ++k) acc = wmma16(a[k], load_b_f16(wt, D_, n0, k * 32), acc);
    int col = n0 + lc;
    float bs = bias[col];
    if (z == 2 && mode == 0) {             // transposed V: contiguous 8 f16 per lane
      int g = m0 / S_, l0 = m0 - g * S_;
      f16* dst = vt + ((size_t)g * D_ + col) * S_ + l0 + rbase;
#pragma unroll
      for (int r = 0; r < 8; ++r) dst[r] = (f16)(acc[r] + bs);
    } else {
      f16* dst = (z == 0) ? qb : (z == 1) ? kb : vb;
#pragma unroll
      for (int r = 0; r < 8; ++r)
        dst[(size_t)(m0 + rbase + r) * D_ + col] = (f16)(acc[r] + bs);
    }
  }
}

// ---------------- time-axis attention: one wave per (qtile, head, group) ----------------
__global__ void k_attn_time(const f16* __restrict__ qb, const f16* __restrict__ kb,
                            const f16* __restrict__ vt, f16* __restrict__ obh) {
  __shared__ float sc[16 * 352];                     // 336 keys padded to 352
  int qt = blockIdx.x, h = blockIdx.y, g = blockIdx.z;
  int lane = threadIdx.x & 31;
  int tok0 = g * S_, q0 = tok0 + qt * 16;
  for (int i = lane; i < 16 * 352; i += 32) sc[i] = 0.f;
  __syncthreads();
  v16h aq = load_a_f16(qb, D_, q0, h * DK_);         // 16x32 Q fragment (DK==WMMA K)
  int rbase = (lane < 16) ? 0 : 8;
  for (int kt = 0; kt < 21; ++kt) {                  // scores = Q * K^T (one WMMA/tile)
    v16h bk_ = load_b_f16(kb + (size_t)tok0 * D_ + h * DK_, D_, kt * 16, 0);
    v8f s = {};
    s = wmma16(aq, bk_, s);
    int col = kt * 16 + (lane & 15);
#pragma unroll
    for (int r = 0; r < 8; ++r) sc[(rbase + r) * 352 + col] = s[r] * SCALE;
  }
  __syncthreads();
  for (int r = 0; r < 16; ++r) {                     // row softmax over 336
    float mx = -3.4e38f;
    for (int i = lane; i < S_; i += 32) mx = fmaxf(mx, sc[r * 352 + i]);
#pragma unroll
    for (int off = 16; off >= 1; off >>= 1) mx = fmaxf(mx, __shfl_xor(mx, off, 32));
    float sum = 0.f;
    for (int i = lane; i < S_; i += 32) {
      float ev = __expf(sc[r * 352 + i] - mx);
      sc[r * 352 + i] = ev; sum += ev;
    }
#pragma unroll
    for (int off = 16; off >= 1; off >>= 1) sum += __shfl_xor(sum, off, 32);
    float inv = 1.f / sum;
    for (int i = lane; i < S_; i += 32) sc[r * 352 + i] *= inv;
  }
  __syncthreads();
  const f16* vtg = vt + (size_t)g * D_ * S_;         // O = P * V  (K padded 336->352, tail==0)
  for (int nh = 0; nh < 2; ++nh) {
    v8f acc = {};
    for (int kt = 0; kt < 11; ++kt)
      acc = wmma16(load_a_f32(sc, 352, 0, kt * 32),
                   load_b_f16(vtg, S_, h * DK_ + nh * 16, kt * 32), acc);
    int col = h * DK_ + nh * 16 + (lane & 15);
#pragma unroll
    for (int r = 0; r < 8; ++r) obh[(size_t)(q0 + rbase + r) * D_ + col] = (f16)acc[r];
  }
}

// ---------------- variable-axis attention (L=8): VALU, one wave per (group, head) ----------------
__global__ void k_attn_var(const f16* __restrict__ qb, const f16* __restrict__ kb,
                           const f16* __restrict__ vb, f16* __restrict__ obh) {
  __shared__ float pm[64];
  int g = blockIdx.x, h = blockIdx.y;
  int lane = threadIdx.x;
  int qi = lane >> 2, k0 = (lane & 3) * 2;
  const f16* qrow = qb + (size_t)(g * 8 + qi) * D_ + h * DK_;
  const f16* kr0  = kb + (size_t)(g * 8 + k0) * D_ + h * DK_;
  const f16* kr1  = kr0 + D_;
  float s0 = 0.f, s1 = 0.f;
#pragma unroll
  for (int d = 0; d < 32; ++d) {
    float q = (float)qrow[d];
    s0 += q * (float)kr0[d];
    s1 += q * (float)kr1[d];
  }
  s0 *= SCALE; s1 *= SCALE;
  float mx = fmaxf(s0, s1);
  mx = fmaxf(mx, __shfl_xor(mx, 1, 4));
  mx = fmaxf(mx, __shfl_xor(mx, 2, 4));
  float e0 = __expf(s0 - mx), e1 = __expf(s1 - mx);
  float sum = e0 + e1;
  sum += __shfl_xor(sum, 1, 4);
  sum += __shfl_xor(sum, 2, 4);
  float inv = 1.f / sum;
  pm[qi * 8 + k0]     = e0 * inv;
  pm[qi * 8 + k0 + 1] = e1 * inv;
  __syncthreads();
  int db = (lane & 3) * 8;
  float acc[8] = {};
#pragma unroll
  for (int k = 0; k < 8; ++k) {
    float p = pm[qi * 8 + k];
    const f16* vr = vb + (size_t)(g * 8 + k) * D_ + h * DK_ + db;
#pragma unroll
    for (int j = 0; j < 8; ++j) acc[j] += p * (float)vr[j];
  }
  f16* orow = obh + (size_t)(g * 8 + qi) * D_ + h * DK_ + db;
#pragma unroll
  for (int j = 0; j < 8; ++j) orow[j] = (f16)acc[j];
}

// ---------------- out-projection + residual + LayerNorm (+optional [B,S,V]->[B,V,S] transpose) ----------------
// Writes both fp32 (residual/LN precision) and f16 shadow (next GEMM A operand).
__global__ void k_outproj_ln(const f16* __restrict__ obh, const f16* __restrict__ wo,
                             const float* __restrict__ bo, const float* __restrict__ resid,
                             const float* __restrict__ g_, const float* __restrict__ b_,
                             float* __restrict__ dst, f16* __restrict__ dsth, int permute) {
  __shared__ float row[16 * 256];
  int m0 = blockIdx.x * 16;
  int wave = threadIdx.x >> 5, lane = threadIdx.x & 31;
  v16h a[8];
#pragma unroll
  for (int k = 0; k < 8; ++k) a[k] = load_a_f16(obh, D_, m0, k * 32);
  int rbase = (lane < 16) ? 0 : 8;
#pragma unroll
  for (int t = 0; t < 4; ++t) {
    int n0 = (wave * 4 + t) * 16;
    v8f acc = {};
#pragma unroll
    for (int k = 0; k < 8; ++k) acc = wmma16(a[k], load_b_f16(wo, D_, n0, k * 32), acc);
    int col = n0 + (lane & 15);
    float bias = bo[col];
#pragma unroll
    for (int r = 0; r < 8; ++r) {
      int mr = rbase + r;
      row[mr * 256 + col] = acc[r] + bias + resid[(size_t)(m0 + mr) * D_ + col];
    }
  }
  __syncthreads();
  int r = threadIdx.x >> 3, sub = threadIdx.x & 7;   // 8 lanes per row (same wave)
  float s = 0.f, s2 = 0.f;
#pragma unroll
  for (int j = 0; j < 32; ++j) {
    float v = row[r * 256 + sub * 32 + j];
    s += v; s2 += v * v;
  }
#pragma unroll
  for (int off = 4; off >= 1; off >>= 1) { s += __shfl_xor(s, off, 8); s2 += __shfl_xor(s2, off, 8); }
  float mean = s * (1.f / 256.f);
  float var  = s2 * (1.f / 256.f) - mean * mean;
  float rstd = rsqrtf(var + LN_EPS);
  int tok = m0 + r;
  size_t drow;
  if (!permute) drow = (size_t)tok;
  else {                                             // [B,S,V] token -> [B,V,S] row
    int b = tok / (S_ * V_);
    int rem = tok - b * S_ * V_;
    int ss = rem / V_, vv = rem - ss * V_;
    drow = (size_t)(b * V_ + vv) * S_ + ss;
  }
#pragma unroll
  for (int j = 0; j < 32; ++j) {
    int c = sub * 32 + j;
    float v = (row[r * 256 + c] - mean) * rstd * g_[c] + b_[c];
    dst[drow * D_ + c]  = v;
    dsth[drow * D_ + c] = (f16)v;
  }
}

// ---------------- norm-LN with [B,V,S]->[B,S,V] transpose (fp32 + f16 shadow) ----------------
__global__ void k_ln_trans(const float* __restrict__ src, const float* __restrict__ g_,
                           const float* __restrict__ b_, float* __restrict__ dst,
                           f16* __restrict__ dsth) {
  int t = blockIdx.x, lane = threadIdx.x;
  const float* rp = src + (size_t)t * D_;
  float v[8];
  float s = 0.f, s2 = 0.f;
#pragma unroll
  for (int j = 0; j < 8; ++j) { v[j] = rp[lane * 8 + j]; s += v[j]; s2 += v[j] * v[j]; }
#pragma unroll
  for (int off = 16; off >= 1; off >>= 1) { s += __shfl_xor(s, off, 32); s2 += __shfl_xor(s2, off, 32); }
  float mean = s * (1.f / 256.f);
  float var  = s2 * (1.f / 256.f) - mean * mean;
  float rstd = rsqrtf(var + LN_EPS);
  int b = t / (V_ * S_);
  int rem = t - b * V_ * S_;
  int vv = rem / S_, ss = rem - vv * S_;
  size_t drow = (size_t)(b * S_ + ss) * V_ + vv;
#pragma unroll
  for (int j = 0; j < 8; ++j) {
    int c = lane * 8 + j;
    float val = (v[j] - mean) * rstd * g_[c] + b_[c];
    dst[drow * D_ + c]  = val;
    dsth[drow * D_ + c] = (f16)val;
  }
}

// ---------------- final projection: split-K WMMA + reduce ----------------
__global__ void k_final_gemm(const f16* __restrict__ e1h, const f16* __restrict__ pwt,
                             float* __restrict__ partial) {
  int ks = blockIdx.x, nt = blockIdx.y;              // 84 K-splits x 6 N-tiles, M=16
  v8f acc = {};
  for (int st = 0; st < 32; ++st) {
    int k0 = ks * 1024 + st * 32;
    acc = wmma16(load_a_f16(e1h, KD, 0, k0), load_b_f16(pwt, KD, nt * 16, k0), acc);
  }
  int lane = threadIdx.x & 31;
  int col = lane & 15, rbase = (lane < 16) ? 0 : 8;
  float* dst = partial + (size_t)(ks * 6 + nt) * 256;
#pragma unroll
  for (int r = 0; r < 8; ++r) dst[(rbase + r) * 16 + col] = acc[r];
}

__global__ void k_final_reduce(const float* __restrict__ partial, const float* __restrict__ pb,
                               float* __restrict__ out) {
  int id = blockIdx.x * blockDim.x + threadIdx.x;
  if (id >= 16 * 96) return;
  int m = id / 96, p = id - m * 96;
  int nt = p >> 4, nc = p & 15;
  float s = 0.f;
  for (int ks = 0; ks < 84; ++ks)
    s += partial[(size_t)(ks * 6 + nt) * 256 + m * 16 + nc];
  s += pb[p];
  int b = m >> 3, c = m & 7;                         // row m = b*V+v, keep all 8 channels
  out[((size_t)b * P_ + p) * 8 + c] = s;
}

// ---------------- launch ----------------
extern "C" void kernel_launch(void* const* d_in, const int* in_sizes, int n_in,
                              void* d_out, int out_size, void* d_ws, size_t ws_size,
                              hipStream_t stream) {
  (void)in_sizes; (void)n_in; (void)out_size; (void)ws_size;
  const float* x     = (const float*)d_in[0];
  const float* emb_w = (const float*)d_in[1];
  const float* emb_b = (const float*)d_in[2];
  const float* Wq    = (const float*)d_in[3];
  const float* bq    = (const float*)d_in[4];
  const float* Wk    = (const float*)d_in[5];
  const float* bk    = (const float*)d_in[6];
  const float* Wv    = (const float*)d_in[7];
  const float* bv    = (const float*)d_in[8];
  const float* Wo    = (const float*)d_in[9];
  const float* bo    = (const float*)d_in[10];
  const float* ln1g  = (const float*)d_in[11];
  const float* ln1b  = (const float*)d_in[12];
  const float* nrmg  = (const float*)d_in[13];
  const float* nrmb  = (const float*)d_in[14];
  const float* projw = (const float*)d_in[15];
  const float* projb = (const float*)d_in[16];
  float* out = (float*)d_out;

  char* cur = (char*)d_ws;
  auto alloc = [&](size_t bytes) {
    char* r = cur;
    cur += (bytes + 255) & ~(size_t)255;
    return r;
  };
  float* e1  = (float*)alloc((size_t)NTOK * D_ * 4);
  float* e2  = (float*)alloc((size_t)NTOK * D_ * 4);
  f16*   e1h = (f16*)alloc((size_t)NTOK * D_ * 2);
  f16*   e2h = (f16*)alloc((size_t)NTOK * D_ * 2);
  f16*   obh = (f16*)alloc((size_t)NTOK * D_ * 2);
  f16* qb = (f16*)alloc((size_t)NTOK * D_ * 2);
  f16* kb = (f16*)alloc((size_t)NTOK * D_ * 2);
  f16* vb = (f16*)alloc((size_t)NTOK * D_ * 2);
  f16* vt = (f16*)alloc((size_t)NTOK * D_ * 2 + 1024);   // +pad for K=352 over-read
  f16* wqt = (f16*)alloc((size_t)D_ * D_ * 2);
  f16* wkt = (f16*)alloc((size_t)D_ * D_ * 2);
  f16* wvt = (f16*)alloc((size_t)D_ * D_ * 2);
  f16* wot = (f16*)alloc((size_t)D_ * D_ * 2);
  f16* pwt = (f16*)alloc((size_t)P_ * KD * 2);
  float* partial = (float*)alloc((size_t)84 * 6 * 256 * 4);

  k_embed<<<NTOK, D_, 0, stream>>>(x, emb_w, emb_b, e1, e1h);
  k_wtrans<<<D_, D_, 0, stream>>>(Wq, wqt);
  k_wtrans<<<D_, D_, 0, stream>>>(Wk, wkt);
  k_wtrans<<<D_, D_, 0, stream>>>(Wv, wvt);
  k_wtrans<<<D_, D_, 0, stream>>>(Wo, wot);
  k_pwtrans<<<KD / 256, 256, 0, stream>>>(projw, pwt);

  for (int it = 0; it < 10; ++it) {
    // time-axis attention on e1 ([B,V,S,D])
    k_qkv<<<NTOK / 16, 128, 0, stream>>>(e1h, wqt, wkt, wvt, bq, bk, bv,
                                         qb, kb, vb, vt, 0);
    k_attn_time<<<dim3(21, H_, B_ * V_), 32, 0, stream>>>(qb, kb, vt, obh);
    k_outproj_ln<<<NTOK / 16, 128, 0, stream>>>(obh, wot, bo, e1, ln1g, ln1b, e1, e1h, 0);
    // norm LN + transpose to [B,S,V,D]
    k_ln_trans<<<NTOK, 32, 0, stream>>>(e1, nrmg, nrmb, e2, e2h);
    // variable-axis attention on e2
    k_qkv<<<NTOK / 16, 128, 0, stream>>>(e2h, wqt, wkt, wvt, bq, bk, bv,
                                         qb, kb, vb, vt, 1);
    k_attn_var<<<dim3(B_ * S_, H_), 32, 0, stream>>>(qb, kb, vb, obh);
    // out-proj + LN, writing transposed back into e1 ([B,V,S,D]) + f16 shadow
    k_outproj_ln<<<NTOK / 16, 128, 0, stream>>>(obh, wot, bo, e2, ln1g, ln1b, e1, e1h, 1);
  }

  k_final_gemm<<<dim3(84, 6), 32, 0, stream>>>(e1h, pwt, partial);
  k_final_reduce<<<6, 256, 0, stream>>>(partial, projb, out);
}
